// TemporalBlock_32220844654807
// MI455X (gfx1250) — matrix operand order repaired and checked by reading, not compiled, not run on hardware
//
#include <hip/hip_runtime.h>

typedef __bf16 bf16_t;
typedef __attribute__((ext_vector_type(16))) __bf16 v16bf;
typedef __attribute__((ext_vector_type(8)))  __bf16 v8bf;
typedef __attribute__((ext_vector_type(8)))  float  v8f;

#define B_    64
#define CIN_  128
#define COUT_ 256
#define L_    1024
#define LP_   1028   // L + causal pad P=4 folded into layout

// ---------------------------------------------------------------------------
// Tensor Data Mover (CDNA5): D# descriptor build + tensor_load_to_lds
// ---------------------------------------------------------------------------
#if defined(__has_builtin)
#if __has_builtin(__builtin_amdgcn_tensor_load_to_lds) && \
    __has_builtin(__builtin_amdgcn_s_wait_tensorcnt)
#define HAVE_TDM 1
#endif
#endif

#ifdef HAVE_TDM
typedef unsigned int u32x4 __attribute__((ext_vector_type(4)));
typedef int          i32x4 __attribute__((ext_vector_type(4)));
typedef int          i32x8 __attribute__((ext_vector_type(8)));

// Load a (tile_z ? 3D : 2D) bf16 tile into LDS via the TDM.
// tile_x = contiguous elements per row, stride_y/stride_z in elements.
// Tensor dims == tile dims (no OOB). data_size = 2 bytes.
__device__ __forceinline__ void tdm_load_tile_bf16(
    const void* lds_dst, const void* gsrc,
    unsigned tile_x, unsigned tile_y, unsigned tile_z,
    unsigned stride_y, unsigned stride_z) {
  unsigned lds_off = (unsigned)(unsigned long long)lds_dst;  // LDS aperture: offset in low 32b
  unsigned long long ga = (unsigned long long)gsrc;
  u32x4 g0;
  g0.x = 1u;                                                   // count=1, user descriptor
  g0.y = lds_off;                                              // lds_addr
  g0.z = (unsigned)(ga & 0xFFFFFFFFu);                         // global_addr[31:0]
  g0.w = (unsigned)((ga >> 32) & 0x1FFFFFFu) | (2u << 30);     // global_addr[56:32] | type=2
  i32x8 g1;
  g1[0] = (int)(1u << 16);                                     // data_size=1 -> 2 bytes
  g1[1] = (int)(tile_x << 16);                                 // tensor_dim0[15:0]   (bits 63:48)
  g1[2] = (int)((tile_x >> 16) | (tile_y << 16));              // dim0[31:16] | dim1[15:0]
  g1[3] = (int)((tile_y >> 16) | (tile_x << 16));              // dim1[31:16] | tile_dim0
  g1[4] = (int)((tile_y & 0xFFFFu) | (tile_z << 16));          // tile_dim1 | tile_dim2
  g1[5] = (int)stride_y;                                       // tensor_dim0_stride[31:0]
  g1[6] = (int)((stride_z & 0xFFFFu) << 16);                   // dim0_str[47:32]=0 | dim1_str[15:0]
  g1[7] = (int)(stride_z >> 16);                               // tensor_dim1_stride[47:16]
  i32x4 g2; g2[0] = (int)(tile_z ? tile_z : 1u); g2[1] = 0; g2[2] = 0; g2[3] = 0;
  i32x4 g3; g3[0] = 0; g3[1] = 0; g3[2] = 0; g3[3] = 0;
#if __clang_major__ >= 23
  i32x8 g4; g4[0]=0; g4[1]=0; g4[2]=0; g4[3]=0; g4[4]=0; g4[5]=0; g4[6]=0; g4[7]=0;
  __builtin_amdgcn_tensor_load_to_lds(g0, g1, g2, g3, g4, 0);
#else
  __builtin_amdgcn_tensor_load_to_lds(g0, g1, g2, g3, 0);
#endif
}
#endif  // HAVE_TDM

// ---------------------------------------------------------------------------
// WMMA helpers (CDNA5 wave32, v_wmma_f32_16x16x32_bf16)
// ---------------------------------------------------------------------------
__device__ __forceinline__ v8f wmma_bf16(v16bf a, v16bf b, v8f c) {
  return __builtin_amdgcn_wmma_f32_16x16x32_bf16(
      /*neg_a=*/false, a, /*neg_b=*/false, b, /*c_mod=*/(short)0, c,
      /*reuse_a=*/false, /*reuse_b=*/false);
}

// A fragment: 16x32 bf16 tile of a row-major MxK matrix.
__device__ __forceinline__ v16bf load_a_frag(const bf16_t* base, int stride,
                                             int m0, int kk, int lrow, int hi) {
  const bf16_t* p = base + (size_t)(m0 + lrow) * stride + kk + hi * 8;
  v8bf lo  = *(const v8bf*)p;
  v8bf hi8 = *(const v8bf*)(p + 16);
  return __builtin_shufflevector(lo, hi8, 0,1,2,3,4,5,6,7,8,9,10,11,12,13,14,15);
}

// B fragment: 32x16 bf16 tile; B stored transposed row-major [n][k].
__device__ __forceinline__ v16bf load_b_frag(const bf16_t* base, int stride,
                                             int n0, int kk, int lrow, int hi) {
  const bf16_t* p = base + (size_t)(n0 + lrow) * stride + kk + hi * 8;
  v8bf lo  = *(const v8bf*)p;
  v8bf hi8 = *(const v8bf*)(p + 16);
  return __builtin_shufflevector(lo, hi8, 0,1,2,3,4,5,6,7,8,9,10,11,12,13,14,15);
}

__device__ __forceinline__ float sigmoid_f(float x) { return 1.f / (1.f + __expf(-x)); }
__device__ __forceinline__ float leaky_f(float x)   { return x > 0.f ? x : 0.2f * x; }

// ---------------------------------------------------------------------------
// Prep kernels
// ---------------------------------------------------------------------------
__global__ void k_prep_x(bf16_t* __restrict__ XTP, const float* __restrict__ x) {
  int i = blockIdx.x * 256 + threadIdx.x;            // over B*LP*CIN
  if (i >= B_ * LP_ * CIN_) return;
  int c  = i & (CIN_ - 1);
  int tp = (i >> 7) % LP_;
  int b  = i / (LP_ * CIN_);
  float v = 0.f;
  if (tp >= 4) v = x[((size_t)b * CIN_ + c) * L_ + (tp - 4)];
  XTP[i] = (bf16_t)v;
}

__global__ void k_f2bf(bf16_t* __restrict__ dst, const float* __restrict__ src, int n) {
  int i = blockIdx.x * 256 + threadIdx.x;
  if (i < n) dst[i] = (bf16_t)src[i];
}

// conv weight [O][I][K] -> [K][O][I] bf16
__global__ void k_convw(bf16_t* __restrict__ dst, const float* __restrict__ src,
                        int O, int I, int Kk) {
  int i = blockIdx.x * 256 + threadIdx.x;
  if (i >= O * I * Kk) return;
  int k  = i % Kk;
  int ii = (i / Kk) % I;
  int o  = i / (Kk * I);
  dst[((size_t)k * O + o) * I + ii] = (bf16_t)src[i];
}

// zero h0 ping-pong buffers, GP causal pad rows, barrier counter
__global__ void k_prep_scan(bf16_t* __restrict__ HBUF, bf16_t* __restrict__ GP,
                            unsigned int* __restrict__ cnt) {
  int i = blockIdx.x * 256 + threadIdx.x;
  if (i == 0) *cnt = 0u;
  if (i < 2 * B_ * COUT_) HBUF[i] = (bf16_t)0.f;
  int j = i - 2 * B_ * COUT_;
  if (j >= 0 && j < B_ * 4 * COUT_) {
    int b = j >> 10;
    int r = j & 1023;
    GP[(size_t)b * LP_ * COUT_ + r] = (bf16_t)0.f;
  }
}

// ---------------------------------------------------------------------------
// Stage 1: conv1 (dilated causal) + LeakyReLU -> H1 bf16 [b][t][cout]
//          and 1x1 skip conv -> d_out f32 [b][cout][t]
// ---------------------------------------------------------------------------
__global__ void __launch_bounds__(256)
k_conv1_skip(const bf16_t* __restrict__ XTP, const bf16_t* __restrict__ W1,
             const bf16_t* __restrict__ SW, const float* __restrict__ c1b,
             const float* __restrict__ skb, bf16_t* __restrict__ H1,
             float* __restrict__ out) {
  __shared__ bf16_t Bs[20][CIN_];              // t0..t0+19 rows of padded x
  int b  = blockIdx.x >> 6;
  int t0 = (blockIdx.x & 63) << 4;
#ifdef HAVE_TDM
  if (threadIdx.x < 32) {
    tdm_load_tile_bf16(&Bs[0][0], &XTP[((size_t)(b * LP_ + t0)) * CIN_],
                       CIN_, 20, 0, CIN_, 0);
    __builtin_amdgcn_s_wait_tensorcnt(0);
  }
#else
  for (int e = threadIdx.x; e < 20 * CIN_ / 8; e += 256) {
    int row = (e * 8) >> 7, col = (e * 8) & (CIN_ - 1);
    *(v8bf*)&Bs[row][col] =
        *(const v8bf*)&XTP[((size_t)(b * LP_ + t0 + row)) * CIN_ + col];
  }
#endif
  __syncthreads();

  int lane = threadIdx.x & 31, wv = threadIdx.x >> 5;
  int lrow = lane & 15, hi = lane >> 4;
  int m0 = wv * 32;
  v8f accC[2] = {v8f{}, v8f{}};
  v8f accS[2] = {v8f{}, v8f{}};

  for (int kk = 0; kk < CIN_; kk += 32) {
    v16bf bf[3];                               // tap k uses x[t - 4 + 2k] -> panel row off 2k
    bf[0] = load_b_frag(&Bs[0][0], CIN_, 0, kk, lrow, hi);
    bf[1] = load_b_frag(&Bs[0][0], CIN_, 2, kk, lrow, hi);
    bf[2] = load_b_frag(&Bs[0][0], CIN_, 4, kk, lrow, hi);
    for (int mt = 0; mt < 2; ++mt) {
      int m = m0 + mt * 16;
      for (int k = 0; k < 3; ++k) {
        v16bf a = load_a_frag(W1 + (size_t)k * COUT_ * CIN_, CIN_, m, kk, lrow, hi);
        accC[mt] = wmma_bf16(a, bf[k], accC[mt]);
      }
      v16bf as = load_a_frag(SW, CIN_, m, kk, lrow, hi);
      accS[mt] = wmma_bf16(as, bf[2], accS[mt]); // skip uses unshifted x (row off 4)
    }
  }

  int t = t0 + lrow;
  for (int mt = 0; mt < 2; ++mt) {
    int cbase = m0 + mt * 16 + hi * 8;
    v8bf hv;
    for (int r = 0; r < 8; ++r) {
      int c = cbase + r;
      float vc = leaky_f(accC[mt][r] + c1b[c]);
      hv[r] = (bf16_t)vc;
      out[((size_t)b * COUT_ + c) * L_ + t] = accS[mt][r] + skb[c];
    }
    *(v8bf*)&H1[((size_t)(b * L_ + t)) * COUT_ + cbase] = hv;
  }
}

// ---------------------------------------------------------------------------
// Stage 2: GRU input projections xz/xr/xh = H1 @ W{z,r,h}^T + bias (f32)
// ---------------------------------------------------------------------------
__global__ void __launch_bounds__(256)
k_proj(const bf16_t* __restrict__ H1,
       const bf16_t* __restrict__ WZ, const bf16_t* __restrict__ WR,
       const bf16_t* __restrict__ WH,
       const float* __restrict__ zb, const float* __restrict__ rb,
       const float* __restrict__ hb,
       float* __restrict__ XZ, float* __restrict__ XR, float* __restrict__ XH) {
  __shared__ bf16_t Bs[16][COUT_];             // 8KB panel of h1
  int b  = blockIdx.x >> 6;
  int t0 = (blockIdx.x & 63) << 4;
#ifdef HAVE_TDM
  if (threadIdx.x < 32) {
    tdm_load_tile_bf16(&Bs[0][0], &H1[((size_t)(b * L_ + t0)) * COUT_],
                       COUT_, 16, 0, COUT_, 0);
    __builtin_amdgcn_s_wait_tensorcnt(0);
  }
#else
  {
    int e = threadIdx.x;
    int row = e >> 4, col = (e & 15) * 16;
    *(v16bf*)&Bs[row][col] =
        *(const v16bf*)&H1[((size_t)(b * L_ + t0 + row)) * COUT_ + col];
  }
#endif
  __syncthreads();

  int lane = threadIdx.x & 31, wv = threadIdx.x >> 5;
  int lrow = lane & 15, hi = lane >> 4;
  int m0 = wv * 32;
  v8f az[2] = {v8f{}, v8f{}}, ar[2] = {v8f{}, v8f{}}, ah[2] = {v8f{}, v8f{}};

  for (int kk = 0; kk < COUT_; kk += 32) {
    v16bf bfrag = load_b_frag(&Bs[0][0], COUT_, 0, kk, lrow, hi);
    for (int mt = 0; mt < 2; ++mt) {
      int m = m0 + mt * 16;
      az[mt] = wmma_bf16(load_a_frag(WZ, COUT_, m, kk, lrow, hi), bfrag, az[mt]);
      ar[mt] = wmma_bf16(load_a_frag(WR, COUT_, m, kk, lrow, hi), bfrag, ar[mt]);
      ah[mt] = wmma_bf16(load_a_frag(WH, COUT_, m, kk, lrow, hi), bfrag, ah[mt]);
    }
  }

  int t = t0 + lrow;
  size_t base = ((size_t)(b * L_ + t)) * COUT_;
  for (int mt = 0; mt < 2; ++mt) {
    int h0 = m0 + mt * 16 + hi * 8;
    for (int r = 0; r < 8; ++r) {
      XZ[base + h0 + r] = az[mt][r] + zb[h0 + r];
      XR[base + h0 + r] = ar[mt][r] + rb[h0 + r];
      XH[base + h0 + r] = ah[mt][r] + hb[h0 + r];
    }
  }
}

// ---------------------------------------------------------------------------
// Stage 3: cooperative GRU scan. 16 blocks x 128 threads (4 waves).
// Block bx owns channel tile n0 = 16*bx; wave w owns batch tile b0 = 16*w.
// UALL = [Uz | Ur | Uh] contiguous; block's 24KB slice loaded once via TDM 3D
// tile and stays LDS-resident for the whole 1024-step scan.
// ---------------------------------------------------------------------------
__global__ void __launch_bounds__(128)
k_gru_scan(const bf16_t* __restrict__ UALL,
           const float* __restrict__ uzb, const float* __restrict__ urb,
           const float* __restrict__ uhb,
           const float* __restrict__ XZ, const float* __restrict__ XR,
           const float* __restrict__ XH, const float* __restrict__ aw,
           bf16_t* __restrict__ HBUF, bf16_t* __restrict__ GP,
           unsigned int* __restrict__ cnt) {
  __shared__ bf16_t Us[3][16][COUT_];          // 24KB: 16 rows of Uz/Ur/Uh
  int n0 = blockIdx.x * 16;
#ifdef HAVE_TDM
  if (threadIdx.x < 32) {
    tdm_load_tile_bf16(&Us[0][0][0], &UALL[(size_t)n0 * COUT_],
                       COUT_, 16, 3, COUT_, COUT_ * COUT_);
    __builtin_amdgcn_s_wait_tensorcnt(0);
  }
#else
  for (int e = threadIdx.x; e < 3 * 16 * COUT_ / 16; e += 128) {
    int mat = e >> 8, row = (e >> 4) & 15, col = (e & 15) * 16;
    *(v16bf*)&Us[mat][row][col] =
        *(const v16bf*)&UALL[(size_t)mat * COUT_ * COUT_ + (size_t)(n0 + row) * COUT_ + col];
  }
#endif
  __syncthreads();

  int lane = threadIdx.x & 31, mb = threadIdx.x >> 5;
  int lrow = lane & 15, hi = lane >> 4;
  int b0 = mb * 16;
  int c = n0 + lrow;
  float thr = 0.1f * sigmoid_f(aw[0]);
  float uzbc = uzb[c], urbc = urb[c], uhbc = uhb[c];
  const unsigned NB = gridDim.x;

  for (int t = 0; t < L_; ++t) {
    const bf16_t* Hc = HBUF + (size_t)(t & 1) * (B_ * COUT_);
    bf16_t*       Hn = HBUF + (size_t)((t + 1) & 1) * (B_ * COUT_);

    v8f az = {}, ar = {}, ah = {};
    for (int kk = 0; kk < COUT_; kk += 32) {
      v16bf a = load_a_frag(Hc, COUT_, b0, kk, lrow, hi);
      az = wmma_bf16(a, load_b_frag(&Us[0][0][0], COUT_, 0, kk, lrow, hi), az);
      ar = wmma_bf16(a, load_b_frag(&Us[1][0][0], COUT_, 0, kk, lrow, hi), ar);
      ah = wmma_bf16(a, load_b_frag(&Us[2][0][0], COUT_, 0, kk, lrow, hi), ah);
    }

    for (int r = 0; r < 8; ++r) {
      int bb = b0 + hi * 8 + r;
      size_t xi = ((size_t)(bb * L_ + t)) * COUT_ + c;
      float z  = sigmoid_f(XZ[xi] + az[r] + uzbc);
      float rr = sigmoid_f(XR[xi] + ar[r] + urbc);
      float ht = tanhf(XH[xi] + rr * (ah[r] + uhbc));
      float hold = (float)Hc[(size_t)bb * COUT_ + c];
      float hn = (1.f - z) * hold + z * ht;
      Hn[(size_t)bb * COUT_ + c] = (bf16_t)hn;           // un-gated h carried
      float g = (fabsf(hn) < thr) ? 0.f : hn;            // attention-zeroed output
      GP[((size_t)(bb * LP_ + t + 4)) * COUT_ + c] = (bf16_t)g;
    }

    // grid-wide barrier (release publish, acquire observe, then L0 invalidate)
    __syncthreads();
    if (threadIdx.x == 0) {
      __hip_atomic_fetch_add(cnt, 1u, __ATOMIC_RELEASE, __HIP_MEMORY_SCOPE_AGENT);
      unsigned target = (unsigned)(t + 1) * NB;
      while (__hip_atomic_load(cnt, __ATOMIC_ACQUIRE, __HIP_MEMORY_SCOPE_AGENT) < target) {
        __builtin_amdgcn_s_sleep(1);
      }
    }
    __syncthreads();
    __threadfence();
  }
}

// ---------------------------------------------------------------------------
// Stage 4: conv2 (dilated causal) + LeakyReLU, accumulate into d_out (skip)
// ---------------------------------------------------------------------------
__global__ void __launch_bounds__(256)
k_conv2_add(const bf16_t* __restrict__ GP, const bf16_t* __restrict__ W2,
            const float* __restrict__ c2b, float* __restrict__ out) {
  __shared__ bf16_t Bs[20][COUT_];             // 10KB
  int b  = blockIdx.x >> 6;
  int t0 = (blockIdx.x & 63) << 4;
#ifdef HAVE_TDM
  if (threadIdx.x < 32) {
    tdm_load_tile_bf16(&Bs[0][0], &GP[((size_t)(b * LP_ + t0)) * COUT_],
                       COUT_, 20, 0, COUT_, 0);
    __builtin_amdgcn_s_wait_tensorcnt(0);
  }
#else
  for (int e = threadIdx.x; e < 20 * COUT_ / 16; e += 256) {
    int row = e >> 4, col = (e & 15) * 16;
    *(v16bf*)&Bs[row][col] =
        *(const v16bf*)&GP[((size_t)(b * LP_ + t0 + row)) * COUT_ + col];
  }
#endif
  __syncthreads();

  int lane = threadIdx.x & 31, wv = threadIdx.x >> 5;
  int lrow = lane & 15, hi = lane >> 4;
  int m0 = wv * 32;
  v8f acc[2] = {v8f{}, v8f{}};

  for (int kk = 0; kk < COUT_; kk += 32) {
    v16bf bf[3];
    bf[0] = load_b_frag(&Bs[0][0], COUT_, 0, kk, lrow, hi);
    bf[1] = load_b_frag(&Bs[0][0], COUT_, 2, kk, lrow, hi);
    bf[2] = load_b_frag(&Bs[0][0], COUT_, 4, kk, lrow, hi);
    for (int mt = 0; mt < 2; ++mt) {
      int m = m0 + mt * 16;
      for (int k = 0; k < 3; ++k) {
        v16bf a = load_a_frag(W2 + (size_t)k * COUT_ * COUT_, COUT_, m, kk, lrow, hi);
        acc[mt] = wmma_bf16(a, bf[k], acc[mt]);
      }
    }
  }

  int t = t0 + lrow;
  for (int mt = 0; mt < 2; ++mt) {
    int cbase = m0 + mt * 16 + hi * 8;
    for (int r = 0; r < 8; ++r) {
      int cc = cbase + r;
      size_t oi = ((size_t)b * COUT_ + cc) * L_ + t;
      out[oi] += leaky_f(acc[mt][r] + c2b[cc]);
    }
  }
}

// ---------------------------------------------------------------------------
// Host launch
// ---------------------------------------------------------------------------
extern "C" void kernel_launch(void* const* d_in, const int* in_sizes, int n_in,
                              void* d_out, int out_size, void* d_ws, size_t ws_size,
                              hipStream_t stream) {
  const float* x   = (const float*)d_in[0];
  const float* c1w = (const float*)d_in[1];
  const float* c1b = (const float*)d_in[2];
  const float* wzw = (const float*)d_in[3];
  const float* wzb = (const float*)d_in[4];
  const float* uzw = (const float*)d_in[5];
  const float* uzb = (const float*)d_in[6];
  const float* wrw = (const float*)d_in[7];
  const float* wrb = (const float*)d_in[8];
  const float* urw = (const float*)d_in[9];
  const float* urb = (const float*)d_in[10];
  const float* whw = (const float*)d_in[11];
  const float* whb = (const float*)d_in[12];
  const float* uhw = (const float*)d_in[13];
  const float* uhb = (const float*)d_in[14];
  const float* c2w = (const float*)d_in[15];
  const float* c2b = (const float*)d_in[16];
  const float* skw = (const float*)d_in[17];
  const float* skb = (const float*)d_in[18];
  const float* aw  = (const float*)d_in[19];
  float* out = (float*)d_out;

  char* ws = (char*)d_ws;
  size_t off = 0;
  auto suballoc = [&](size_t bytes) {
    void* p = ws + off;
    off = (off + bytes + 255) & ~(size_t)255;
    return p;
  };
  unsigned int* cnt = (unsigned int*)suballoc(256);
  bf16_t* W1   = (bf16_t*)suballoc((size_t)3 * COUT_ * CIN_ * 2);
  bf16_t* SW   = (bf16_t*)suballoc((size_t)COUT_ * CIN_ * 2);
  bf16_t* WZ   = (bf16_t*)suballoc((size_t)COUT_ * COUT_ * 2);
  bf16_t* WR   = (bf16_t*)suballoc((size_t)COUT_ * COUT_ * 2);
  bf16_t* WH   = (bf16_t*)suballoc((size_t)COUT_ * COUT_ * 2);
  bf16_t* UALL = (bf16_t*)suballoc((size_t)3 * COUT_ * COUT_ * 2);  // Uz|Ur|Uh contiguous
  bf16_t* W2   = (bf16_t*)suballoc((size_t)3 * COUT_ * COUT_ * 2);
  bf16_t* HBUF = (bf16_t*)suballoc((size_t)2 * B_ * COUT_ * 2);
  bf16_t* XTP  = (bf16_t*)suballoc((size_t)B_ * LP_ * CIN_ * 2);
  bf16_t* H1   = (bf16_t*)suballoc((size_t)B_ * L_ * COUT_ * 2);
  bf16_t* GP   = (bf16_t*)suballoc((size_t)B_ * LP_ * COUT_ * 2);
  float*  XZ   = (float*)suballoc((size_t)B_ * L_ * COUT_ * 4);
  float*  XR   = (float*)suballoc((size_t)B_ * L_ * COUT_ * 4);
  float*  XH   = (float*)suballoc((size_t)B_ * L_ * COUT_ * 4);

  const int NU = COUT_ * COUT_;

  // --- prep ---
  {
    int n = B_ * LP_ * CIN_;
    k_prep_x<<<(n + 255) / 256, 256, 0, stream>>>(XTP, x);
  }
  k_f2bf<<<(NU + 255) / 256, 256, 0, stream>>>(WZ, wzw, NU);
  k_f2bf<<<(NU + 255) / 256, 256, 0, stream>>>(WR, wrw, NU);
  k_f2bf<<<(NU + 255) / 256, 256, 0, stream>>>(WH, whw, NU);
  k_f2bf<<<(NU + 255) / 256, 256, 0, stream>>>(UALL + 0 * NU, uzw, NU);
  k_f2bf<<<(NU + 255) / 256, 256, 0, stream>>>(UALL + 1 * NU, urw, NU);
  k_f2bf<<<(NU + 255) / 256, 256, 0, stream>>>(UALL + 2 * NU, uhw, NU);
  k_f2bf<<<(COUT_ * CIN_ + 255) / 256, 256, 0, stream>>>(SW, skw, COUT_ * CIN_);
  k_convw<<<(3 * COUT_ * CIN_ + 255) / 256, 256, 0, stream>>>(W1, c1w, COUT_, CIN_, 3);
  k_convw<<<(3 * NU + 255) / 256, 256, 0, stream>>>(W2, c2w, COUT_, COUT_, 3);
  {
    int n = 2 * B_ * COUT_ + B_ * 4 * COUT_;
    k_prep_scan<<<(n + 255) / 256, 256, 0, stream>>>(HBUF, GP, cnt);
  }

  // --- pipeline ---
  dim3 gtiles(B_ * (L_ / 16));  // 4096
  k_conv1_skip<<<gtiles, 256, 0, stream>>>(XTP, W1, SW, c1b, skb, H1, out);
  k_proj<<<gtiles, 256, 0, stream>>>(H1, WZ, WR, WH, wzb, wrb, whb, XZ, XR, XH);
  k_gru_scan<<<16, 128, 0, stream>>>(UALL, uzb, urb, uhb, XZ, XR, XH, aw,
                                     HBUF, GP, cnt);
  k_conv2_add<<<gtiles, 256, 0, stream>>>(GP, W2, c2b, out);
}
